// MultiHeadSelfAttention_64476049047756
// MI455X (gfx1250) — matrix operand order repaired and checked by reading, not compiled, not run on hardware
//
#include <hip/hip_runtime.h>

typedef __attribute__((ext_vector_type(16))) __bf16        v16bf;
typedef __attribute__((ext_vector_type(8)))  float         v8f;
typedef __attribute__((ext_vector_type(4)))  unsigned int  u32x4;

constexpr int Bn  = 4;
constexpr int Sn  = 2048;
constexpr int Dn  = 1024;
constexpr int Hn  = 16;
constexpr int DKn = 64;
constexpr int BSn = Bn * Sn;   // 8192 rows

union F16Frag { u32x4 u[2]; v16bf v; };

// ---------------------------------------------------------------------------
// A-fragment (16x32 bf16, MxK) per ISA layout:
//   lanes 0-15: row M = lane,     elems 0-7 -> K=k0..k0+7,   elems 8-15 -> K=k0+16..k0+23
//   lanes 16-31: row M = lane-16, elems 0-7 -> K=k0+8..k0+15, elems 8-15 -> K=k0+24..k0+31
__device__ __forceinline__ v16bf load_A_frag(const __bf16* base, int ld, int m0, int k0, int lane) {
  int m  = m0 + (lane & 15);
  int kb = k0 + ((lane >> 4) << 3);
  const __bf16* p = base + (size_t)m * ld + kb;
  F16Frag f;
  f.u[0] = *(const u32x4*)(p);
  f.u[1] = *(const u32x4*)(p + 16);
  return f.v;
}

// B-fragment (32x16 bf16, KxN) with B[k][n] = base[n][k] (transposed source):
//   lane -> column N = n0 + (lane&15); elems 0-7 -> K=k0+8*(lane>>4)+0..7, elems 8-15 -> K=k0+16+8*(lane>>4)+0..7.
__device__ __forceinline__ v16bf load_Bt_frag(const __bf16* base, int ld, int n0, int k0, int lane) {
  int n  = n0 + (lane & 15);
  int kb = k0 + ((lane >> 4) << 3);
  const __bf16* p = base + (size_t)n * ld + kb;
  F16Frag f;
  f.u[0] = *(const u32x4*)(p);
  f.u[1] = *(const u32x4*)(p + 16);
  return f.v;
}

__device__ __forceinline__ v8f wmma_bf16(v16bf a, v16bf b, v8f c) {
  return __builtin_amdgcn_wmma_f32_16x16x32_bf16(false, a, false, b, (short)0, c, false, false);
}

// ---------------------------------------------------------------------------
__global__ void f32_to_bf16_kernel(const float* __restrict__ src, __bf16* __restrict__ dst, int n) {
  int i = blockIdx.x * blockDim.x + threadIdx.x;
  if (i < n) dst[i] = (__bf16)src[i];
}

// ---------------------------------------------------------------------------
// C[M,N] = A[M,K] * W[N,K]^T ;  block = 8 waves (2x4) -> 128x128 tile,
// wave -> 64x32 (8 accumulators: 8 WMMAs per 6 fragment loads).
__global__ __launch_bounds__(256) void gemm_nt_bf16(const __bf16* __restrict__ A,
                                                    const __bf16* __restrict__ W,
                                                    __bf16* __restrict__ Cb,
                                                    float*  __restrict__ Cf,
                                                    int M, int N, int K, int write_f32) {
  int lane = threadIdx.x & 31;
  int wave = threadIdx.x >> 5;
  int m0 = blockIdx.y * 128 + (wave & 1) * 64;
  int n0 = blockIdx.x * 128 + (wave >> 1) * 32;
  v8f acc[4][2] = {};
#pragma unroll 2
  for (int k0 = 0; k0 < K; k0 += 32) {
    v16bf a0 = load_A_frag(A, K, m0,      k0, lane);
    v16bf a1 = load_A_frag(A, K, m0 + 16, k0, lane);
    v16bf a2 = load_A_frag(A, K, m0 + 32, k0, lane);
    v16bf a3 = load_A_frag(A, K, m0 + 48, k0, lane);
    v16bf b0 = load_Bt_frag(W, K, n0,      k0, lane);
    v16bf b1 = load_Bt_frag(W, K, n0 + 16, k0, lane);
    acc[0][0] = wmma_bf16(a0, b0, acc[0][0]);
    acc[0][1] = wmma_bf16(a0, b1, acc[0][1]);
    acc[1][0] = wmma_bf16(a1, b0, acc[1][0]);
    acc[1][1] = wmma_bf16(a1, b1, acc[1][1]);
    acc[2][0] = wmma_bf16(a2, b0, acc[2][0]);
    acc[2][1] = wmma_bf16(a2, b1, acc[2][1]);
    acc[3][0] = wmma_bf16(a3, b0, acc[3][0]);
    acc[3][1] = wmma_bf16(a3, b1, acc[3][1]);
  }
  int rb = (lane >> 4) << 3;
  int cc = lane & 15;
#pragma unroll
  for (int mi = 0; mi < 4; ++mi) {
#pragma unroll
    for (int r = 0; r < 8; ++r) {
      size_t rw = (size_t)(m0 + mi * 16 + rb + r) * N;
      if (write_f32) {
        Cf[rw + n0 + cc]      = acc[mi][0][r];
        Cf[rw + n0 + 16 + cc] = acc[mi][1][r];
      } else {
        Cb[rw + n0 + cc]      = (__bf16)acc[mi][0][r];
        Cb[rw + n0 + 16 + cc] = (__bf16)acc[mi][1][r];
      }
    }
  }
}

// ---------------------------------------------------------------------------
// [b,s,h,dk] -> [b,h,s,dk], optional interleaved RoPE (pairs 2i,2i+1)
__global__ void rope_transpose(const __bf16* __restrict__ in, __bf16* __restrict__ out,
                               const int* __restrict__ pos, int do_rope) {
  int idx = blockIdx.x * blockDim.x + threadIdx.x;
  if (idx >= Bn * Sn * Hn * (DKn / 2)) return;
  int p  = idx & 31;                 // pair index 0..31
  int h  = (idx >> 5) & (Hn - 1);
  int si = (idx >> 9) & (Sn - 1);
  int bi = idx >> 20;
  const __bf16* src = in + ((((size_t)bi * Sn + si) * Hn + h) * DKn) + 2 * p;
  float x1 = (float)src[0];
  float x2 = (float)src[1];
  if (do_rope) {
    // inv_freq = theta^(-2p/64), theta=10000
    float inv = expf(-(float)(2 * p) * (9.210340371976184f / 64.0f));
    float ang = (float)pos[si] * inv;
    float cs = cosf(ang);
    float sn = sinf(ang);
    float r1 = x1 * cs - x2 * sn;
    float r2 = x1 * sn + x2 * cs;
    x1 = r1; x2 = r2;
  }
  __bf16* dst = out + ((((size_t)bi * Hn + h) * Sn + si) * DKn) + 2 * p;
  dst[0] = (__bf16)x1;
  dst[1] = (__bf16)x2;
}

// ---------------------------------------------------------------------------
// Wave-level flash attention. Grid: (S/64, H, B). Block: 128 thr = 4 waves,
// each wave owns 16 query rows. Causal, online softmax in fp32.
// K tile is staged with CDNA5 async global->LDS copies (ASYNCcnt).
__global__ __launch_bounds__(128) void flash_attn(const __bf16* __restrict__ Q,
                                                  const __bf16* __restrict__ Kh,
                                                  const __bf16* __restrict__ Vh,
                                                  __bf16* __restrict__ O) {
  __shared__ __attribute__((aligned(16))) __bf16 Kt [32 * DKn];   // [key][d]
  __shared__ __attribute__((aligned(16))) __bf16 VtT[DKn * 32];   // [d][key]
  __shared__ __attribute__((aligned(16))) __bf16 Pb [4][16 * 32]; // per-wave P patch

  int lane = threadIdx.x & 31;
  int wave = threadIdx.x >> 5;
  int tid  = threadIdx.x;
  int bx = blockIdx.x, hy = blockIdx.y, bz = blockIdx.z;

  size_t headoff = ((size_t)bz * Hn + hy) * (size_t)Sn * DKn;
  const __bf16* Qb = Q  + headoff;
  const __bf16* Kb = Kh + headoff;
  const __bf16* Vb = Vh + headoff;

  int qb = bx * 64 + wave * 16;
  v16bf aq0 = load_A_frag(Qb, DKn, qb, 0,  lane);
  v16bf aq1 = load_A_frag(Qb, DKn, qb, 32, lane);

  v8f o0 = {}, o1 = {}, o2 = {}, o3 = {};
  float mrow[8], lrow[8];
#pragma unroll
  for (int r = 0; r < 8; ++r) { mrow[r] = -1e30f; lrow[r] = 0.0f; }
  const float scale = 0.125f;  // 1/sqrt(64)

  int rb = (lane >> 4) << 3;
  int cc = lane & 15;
  int jend = bx * 64 + 64;

  // per-thread staging coordinates (16 elements each for K and V tiles)
  int sidx  = tid * 16;            // 128*16 = 2048 = 32*64 elements
  int srow  = sidx >> 6;
  int scol  = sidx & 63;
  unsigned ktLds = (unsigned)(size_t)(&Kt[sidx]);   // LDS byte address (low 32 bits)

  for (int j0 = 0; j0 < jend; j0 += 32) {
    // --- K tile: async global->LDS, 2 x b128 per thread (offset applies to both sides)
    {
      const __bf16* ksrc = Kb + (size_t)(j0 + srow) * DKn + scol;
      asm volatile(
          "global_load_async_to_lds_b128 %0, %1, off\n\t"
          "global_load_async_to_lds_b128 %0, %1, off offset:16"
          :: "v"(ktLds), "v"(ksrc) : "memory");
      // --- V tile: transposed scatter into LDS ([d][key]) ---
      const __bf16* vs = Vb + (size_t)(j0 + srow) * DKn + scol;
#pragma unroll
      for (int e = 0; e < 16; ++e) VtT[(scol + e) * 32 + srow] = vs[e];
      // drain this wave's async copies before the block-wide barrier
      asm volatile("s_wait_asynccnt 0" ::: "memory");
    }
    __syncthreads();

    // --- S = Q * K^T (16x32 per wave) ---
    v8f s0 = {}, s1 = {};
    s0 = wmma_bf16(aq0, load_Bt_frag(Kt, DKn, 0,  0,  lane), s0);
    s0 = wmma_bf16(aq1, load_Bt_frag(Kt, DKn, 0,  32, lane), s0);
    s1 = wmma_bf16(aq0, load_Bt_frag(Kt, DKn, 16, 0,  lane), s1);
    s1 = wmma_bf16(aq1, load_Bt_frag(Kt, DKn, 16, 32, lane), s1);

    __bf16* Pw = &Pb[wave][0];
#pragma unroll
    for (int r = 0; r < 8; ++r) {
      int row = qb + rb + r;
      float e0 = s0[r] * scale;
      float e1 = s1[r] * scale;
      if (j0 + cc > row)      e0 = -__builtin_inff();
      if (j0 + 16 + cc > row) e1 = -__builtin_inff();
      float mx = fmaxf(e0, e1);
      mx = fmaxf(mx, __shfl_xor(mx, 1, 32));
      mx = fmaxf(mx, __shfl_xor(mx, 2, 32));
      mx = fmaxf(mx, __shfl_xor(mx, 4, 32));
      mx = fmaxf(mx, __shfl_xor(mx, 8, 32));
      float mnew = fmaxf(mrow[r], mx);
      float corr = __expf(mrow[r] - mnew);
      float p0 = __expf(e0 - mnew);
      float p1 = __expf(e1 - mnew);
      float ps = p0 + p1;
      ps += __shfl_xor(ps, 1, 32);
      ps += __shfl_xor(ps, 2, 32);
      ps += __shfl_xor(ps, 4, 32);
      ps += __shfl_xor(ps, 8, 32);
      lrow[r] = lrow[r] * corr + ps;
      mrow[r] = mnew;
      o0[r] *= corr; o1[r] *= corr; o2[r] *= corr; o3[r] *= corr;
      Pw[(rb + r) * 32 + cc]      = (__bf16)p0;
      Pw[(rb + r) * 32 + 16 + cc] = (__bf16)p1;
    }
    // same-wave DS ops are in-order; fence keeps the compiler from reordering
    asm volatile("s_wait_dscnt 0" ::: "memory");

    // --- O += P * V ---
    v16bf ap = load_A_frag(Pw, 32, 0, 0, lane);
    o0 = wmma_bf16(ap, load_Bt_frag(VtT, 32, 0,  0, lane), o0);
    o1 = wmma_bf16(ap, load_Bt_frag(VtT, 32, 16, 0, lane), o1);
    o2 = wmma_bf16(ap, load_Bt_frag(VtT, 32, 32, 0, lane), o2);
    o3 = wmma_bf16(ap, load_Bt_frag(VtT, 32, 48, 0, lane), o3);
    __syncthreads();
  }

  // --- normalize and write O as [b, s, h*dk] bf16 ---
#pragma unroll
  for (int r = 0; r < 8; ++r) {
    int row = qb + rb + r;
    float inv_l = 1.0f / lrow[r];
    size_t base = ((size_t)bz * Sn + row) * Dn + (size_t)hy * DKn + cc;
    O[base + 0]  = (__bf16)(o0[r] * inv_l);
    O[base + 16] = (__bf16)(o1[r] * inv_l);
    O[base + 32] = (__bf16)(o2[r] * inv_l);
    O[base + 48] = (__bf16)(o3[r] * inv_l);
  }
}

// ---------------------------------------------------------------------------
extern "C" void kernel_launch(void* const* d_in, const int* in_sizes, int n_in,
                              void* d_out, int out_size, void* d_ws, size_t ws_size,
                              hipStream_t stream) {
  const float* x   = (const float*)d_in[0];
  const float* Wq  = (const float*)d_in[1];
  const float* Wk  = (const float*)d_in[2];
  const float* Wv  = (const float*)d_in[3];
  const float* Wo  = (const float*)d_in[4];
  const int*   pos = (const int*)d_in[5];

  char* ws = (char*)d_ws;
  size_t off = 0;
  auto alloc = [&](size_t bytes) -> char* {
    char* p = ws + off;
    off = (off + bytes + 255) & ~(size_t)255;
    return p;
  };

  __bf16* Xb  = (__bf16*)alloc((size_t)BSn * Dn * 2);
  __bf16* Wqb = (__bf16*)alloc((size_t)Dn * Dn * 2);
  __bf16* Wkb = (__bf16*)alloc((size_t)Dn * Dn * 2);
  __bf16* Wvb = (__bf16*)alloc((size_t)Dn * Dn * 2);
  __bf16* Wob = (__bf16*)alloc((size_t)Dn * Dn * 2);
  __bf16* Qt  = (__bf16*)alloc((size_t)BSn * Dn * 2);   // [b,s,h,dk]
  __bf16* Kt2 = (__bf16*)alloc((size_t)BSn * Dn * 2);
  __bf16* Vt2 = (__bf16*)alloc((size_t)BSn * Dn * 2);
  __bf16* Qr  = (__bf16*)alloc((size_t)BSn * Dn * 2);   // [b,h,s,dk]
  __bf16* Kr  = (__bf16*)alloc((size_t)BSn * Dn * 2);
  __bf16* Vr  = (__bf16*)alloc((size_t)BSn * Dn * 2);
  __bf16* Ob  = Xb;  // x is dead after the QKV GEMMs -> reuse for attention out

  // 1) fp32 -> bf16
  int nx = BSn * Dn;
  f32_to_bf16_kernel<<<(nx + 255) / 256, 256, 0, stream>>>(x, Xb, nx);
  int nw = Dn * Dn;
  f32_to_bf16_kernel<<<(nw + 255) / 256, 256, 0, stream>>>(Wq, Wqb, nw);
  f32_to_bf16_kernel<<<(nw + 255) / 256, 256, 0, stream>>>(Wk, Wkb, nw);
  f32_to_bf16_kernel<<<(nw + 255) / 256, 256, 0, stream>>>(Wv, Wvb, nw);
  f32_to_bf16_kernel<<<(nw + 255) / 256, 256, 0, stream>>>(Wo, Wob, nw);

  // 2) Q/K/V projections (WMMA)
  dim3 gg(Dn / 128, BSn / 128);
  gemm_nt_bf16<<<gg, 256, 0, stream>>>(Xb, Wqb, Qt,  nullptr, BSn, Dn, Dn, 0);
  gemm_nt_bf16<<<gg, 256, 0, stream>>>(Xb, Wkb, Kt2, nullptr, BSn, Dn, Dn, 0);
  gemm_nt_bf16<<<gg, 256, 0, stream>>>(Xb, Wvb, Vt2, nullptr, BSn, Dn, Dn, 0);

  // 3) RoPE + head-major transpose
  int nr = Bn * Sn * Hn * (DKn / 2);
  rope_transpose<<<(nr + 255) / 256, 256, 0, stream>>>(Qt,  Qr, pos, 1);
  rope_transpose<<<(nr + 255) / 256, 256, 0, stream>>>(Kt2, Kr, pos, 1);
  rope_transpose<<<(nr + 255) / 256, 256, 0, stream>>>(Vt2, Vr, pos, 0);

  // 4) causal flash attention (WMMA + async global->LDS staging)
  flash_attn<<<dim3(Sn / 64, Hn, Bn), 128, 0, stream>>>(Qr, Kr, Vr, Ob);

  // 5) output projection -> fp32 d_out (WMMA)
  gemm_nt_bf16<<<gg, 256, 0, stream>>>(Ob, Wob, nullptr, (float*)d_out, BSn, Dn, Dn, 1);
}
